// InstanceSegmentationNet2_68436008894699
// MI455X (gfx1250) — matrix operand order repaired and checked by reading, not compiled, not run on hardware
//
#include <hip/hip_runtime.h>
#include <hip/hip_bf16.h>

typedef __attribute__((ext_vector_type(16))) _Float16 v16h;
typedef __attribute__((ext_vector_type(8)))  _Float16 v8h;
typedef __attribute__((ext_vector_type(4)))  _Float16 v4h;
typedef __attribute__((ext_vector_type(8)))  float    v8f;

// ---------------------------------------------------------------------------
// FPS: one block per batch; dist array lives in LDS; argmax with first-index
// tie-break (matches jnp.argmax).
// ---------------------------------------------------------------------------
__global__ __launch_bounds__(256) void k_fps(const float* __restrict__ pts, int ldp,
                                             int Np, int M, int* __restrict__ outIdx)
{
    __shared__ float dist[8192];
    __shared__ float redv[256];
    __shared__ int   redi[256];
    __shared__ int   s_last;
    const int t = threadIdx.x;
    for (int i = t; i < Np; i += 256) dist[i] = 1e10f;
    if (t == 0) s_last = 0;
    __syncthreads();
    for (int it = 0; it < M; ++it) {
        const int last = s_last;
        if (t == 0) outIdx[it] = last;
        const float lx = pts[last], ly = pts[ldp + last], lz = pts[2 * ldp + last];
        float best = -1.0f; int bi = 0x7fffffff;
        for (int i = t; i < Np; i += 256) {
            float dx = pts[i] - lx;
            float dy = pts[ldp + i] - ly;
            float dz = pts[2 * ldp + i] - lz;
            float d  = dx * dx + dy * dy + dz * dz;
            float nd = fminf(dist[i], d);
            dist[i] = nd;
            if (nd > best) { best = nd; bi = i; }
        }
        redv[t] = best; redi[t] = bi;
        __syncthreads();
        for (int s = 128; s > 0; s >>= 1) {
            if (t < s) {
                float ov = redv[t + s]; int oi = redi[t + s];
                if (ov > redv[t] || (ov == redv[t] && oi < redi[t])) { redv[t] = ov; redi[t] = oi; }
            }
            __syncthreads();
        }
        if (t == 0) s_last = redi[0];
        __syncthreads();
    }
}

__global__ __launch_bounds__(256) void k_gather_centers(const float* __restrict__ pts, int ldp,
                                                        const int* __restrict__ idx, int M,
                                                        float* __restrict__ centers)
{
    int i = blockIdx.x * blockDim.x + threadIdx.x;  // over 3*M
    if (i >= 3 * M) return;
    int ch = i / M, j = i % M;
    centers[ch * M + j] = pts[ch * ldp + idx[j]];
}

// Ball query: first K points (in index order) with d2 < r2; pad with first hit
// (or 0 if no hits).
__global__ __launch_bounds__(256) void k_ball(const float* __restrict__ centers, int M,
                                              const float* __restrict__ pts, int ldp, int Np,
                                              float r2, int K, int* __restrict__ nidx)
{
    int j = blockIdx.x * blockDim.x + threadIdx.x;
    if (j >= M) return;
    float cx = centers[j], cy = centers[M + j], cz = centers[2 * M + j];
    int cnt = 0;
    int* out = nidx + (size_t)j * K;
    for (int i = 0; i < Np && cnt < K; ++i) {
        float dx = pts[i] - cx, dy = pts[ldp + i] - cy, dz = pts[2 * ldp + i] - cz;
        if (dx * dx + dy * dy + dz * dz < r2) out[cnt++] = i;
    }
    int pad = (cnt > 0) ? out[0] : 0;
    for (int k = cnt; k < K; ++k) out[k] = pad;
}

// Build SA input (f16 out): ch<3 -> relative coords, ch>=3 -> gathered features.
// TF = feature source type (float for raw input, _Float16 for our buffers).
template <typename TF>
__global__ __launch_bounds__(256) void k_gather_sa(_Float16* __restrict__ out, int cols, int K,
                                                   const int* __restrict__ nidx,
                                                   const float* __restrict__ coords, int ldc,
                                                   const float* __restrict__ centers, int M,
                                                   const TF* __restrict__ feats, int ldf)
{
    int q = blockIdx.x * blockDim.x + threadIdx.x;
    if (q >= cols) return;
    int ch = blockIdx.y;
    int n = nidx[q];
    int j = q / K;
    float v;
    if (ch < 3) v = coords[ch * ldc + n] - centers[ch * M + j];
    else        v = (float)feats[(size_t)(ch - 3) * ldf + n];
    out[(size_t)ch * cols + q] = (_Float16)v;
}

// ---------------------------------------------------------------------------
// One-time per-launch parameter prep: convert W to f16 padded to coP x cinP,
// and pack [bias | g | beta] padded to coP (0 / 1 / 0 in the pad region).
// ---------------------------------------------------------------------------
__global__ __launch_bounds__(256) void k_cvt_w(const float* __restrict__ W, int co, int cin,
                                               _Float16* __restrict__ Wh, int coP, int cinP)
{
    int i = blockIdx.x * blockDim.x + threadIdx.x;
    if (i >= coP * cinP) return;
    int r = i / cinP, k = i % cinP;
    float v = (r < co && k < cin) ? W[(size_t)r * cin + k] : 0.0f;
    Wh[i] = (_Float16)v;
}

__global__ __launch_bounds__(256) void k_pad_params(const float* __restrict__ b,
                                                    const float* __restrict__ g,
                                                    const float* __restrict__ be,
                                                    int co, int coP, float* __restrict__ out)
{
    int i = blockIdx.x * blockDim.x + threadIdx.x;
    if (i >= coP) return;
    out[i]           = (i < co) ? b[i] : 0.0f;
    out[coP + i]     = (g  && i < co) ? g[i]  : 1.0f;
    out[2 * coP + i] = (be && i < co) ? be[i] : 0.0f;
}

__global__ __launch_bounds__(256) void k_zeroh(_Float16* __restrict__ p, int n)
{
    int i = blockIdx.x * blockDim.x + threadIdx.x;
    if (i < n) p[i] = (_Float16)0.0f;
}

// ---------------------------------------------------------------------------
// Double-buffered LDS-tiled WMMA GEMM, f16 activations in / TOUT out:
//   Y[co x cols] = epilogue(Wh[coP x cinP](f16) * X[cinP x cols](f16)).
// Block (8 waves) computes a 64x32 tile as 4x2 wave tiles of 16x16, BK=32.
// Ping-pong LDS tiles -> exactly ONE barrier per K-step; global prefetch for
// step i+1 issues before the barrier, overlapping the WMMA. Guard-free:
// cols % 32 == 0; coP % 64 == 0; cinP % 32 == 0; X pad rows zero-filled.
// ---------------------------------------------------------------------------
#define GEMM_BM 64
#define GEMM_BN 32
#define GEMM_BK 32
#define LDS_STR 40   // halves per row (80B stride: conflict-free b128 reads)

template <typename TOUT>
__global__ __launch_bounds__(256) void k_gemm_wmma(const _Float16* __restrict__ Wh,
                                                   const float* __restrict__ pv,
                                                   const _Float16* __restrict__ X,
                                                   TOUT* __restrict__ Y,
                                                   int co, int coP, int cinP, int cols,
                                                   int relu)
{
    alignas(16) __shared__ _Float16 As[2][GEMM_BM * LDS_STR];   // [row][k]
    alignas(16) __shared__ _Float16 Bs[2][GEMM_BN * LDS_STR];   // [col][k] (X^T)

    const int t = threadIdx.x;
    const int tilesN = cols >> 5;
    const int row0 = (blockIdx.x / tilesN) * GEMM_BM;
    const int col0 = (blockIdx.x % tilesN) * GEMM_BN;

    const int wave = t >> 5;
    const int lane = t & 31;
    const int wrow = (wave & 3) * 16;
    const int wcol = (wave >> 2) * 16;
    const int hi = lane >> 4;
    const int l  = lane & 15;

    // staging assignments
    const int ar = t >> 2;          // 0..63  (A row)
    const int ak = (t & 3) * 8;     // 0,8,16,24 (A k-subgroup, 8 halves = 16B)
    const int bk = t >> 3;          // 0..31  (B k)
    const int bc = (t & 7) * 4;     // 0..28  (B col group of 4, 8B)
    const _Float16* wsrc = Wh + (size_t)(row0 + ar) * cinP + ak;
    const _Float16* xsrc = X + (size_t)bk * cols + col0 + bc;

    v8f acc = {};
    auto compute = [&](int sel) {
        v16h a, b;
        const _Float16* ap = &As[sel][(wrow + l) * LDS_STR + hi * 8];
#pragma unroll
        for (int h = 0; h < 8; ++h) { a[h] = ap[h]; a[h + 8] = ap[h + 16]; }
        const _Float16* bp = &Bs[sel][(wcol + l) * LDS_STR + hi * 8];
#pragma unroll
        for (int h = 0; h < 8; ++h) { b[h] = bp[h]; b[h + 8] = bp[h + 16]; }
        acc = __builtin_amdgcn_wmma_f32_16x16x32_f16(false, a, false, b, (short)0, acc,
                                                     false, false);
    };
    auto stage = [&](int sel, v8h av, v4h xv) {
        *(v8h*)(&As[sel][ar * LDS_STR + ak]) = av;
        Bs[sel][(bc + 0) * LDS_STR + bk] = xv[0];
        Bs[sel][(bc + 1) * LDS_STR + bk] = xv[1];
        Bs[sel][(bc + 2) * LDS_STR + bk] = xv[2];
        Bs[sel][(bc + 3) * LDS_STR + bk] = xv[3];
    };

    // prologue: stage tile 0
    stage(0, *(const v8h*)wsrc, *(const v4h*)xsrc);
    int cur = 0;
    for (int k0 = GEMM_BK; k0 < cinP; k0 += GEMM_BK) {
        v8h av = *(const v8h*)(wsrc + k0);                       // prefetch tile k0
        v4h xv = *(const v4h*)(xsrc + (size_t)k0 * cols);
        __syncthreads();                                         // buf[cur] visible
        compute(cur);                                            // consume tile k0-32
        stage(cur ^ 1, av, xv);                                  // stage tile k0
        cur ^= 1;
    }
    __syncthreads();
    compute(cur);                                                // last tile

#pragma unroll
    for (int v = 0; v < 8; ++v) {
        int r = row0 + wrow + v + hi * 8;        // C/D: VGPR v, hi lanes -> M+8
        int c = col0 + wcol + l;
        float y = acc[v] + pv[r];
        if (relu) { y = y * pv[coP + r] + pv[2 * coP + r]; y = fmaxf(y, 0.0f); }
        if (r < co) Y[(size_t)r * cols + c] = (TOUT)y;
    }
}

// max over K neighbors: in C x (M*K) f16 -> out C x M f16
__global__ __launch_bounds__(256) void k_maxpool_k(const _Float16* __restrict__ in, int M,
                                                   int K, int C, _Float16* __restrict__ out)
{
    int i = blockIdx.x * blockDim.x + threadIdx.x;
    if (i >= C * M) return;
    int ch = i / M, j = i % M;
    const _Float16* p = in + (size_t)ch * M * K + (size_t)j * K;
    float m = (float)p[0];
    for (int k = 1; k < K; ++k) m = fmaxf(m, (float)p[k]);
    out[(size_t)ch * M + j] = (_Float16)m;
}

// global max over columns (sa2): out[ch] = max_c in[ch, c]; f16 -> f32
__global__ __launch_bounds__(256) void k_maxcols(const _Float16* __restrict__ in, int C,
                                                 int cols, float* __restrict__ out)
{
    int ch = blockIdx.x * blockDim.x + threadIdx.x;
    if (ch >= C) return;
    const _Float16* p = in + (size_t)ch * cols;
    float m = (float)p[0];
    for (int c = 1; c < cols; ++c) m = fmaxf(m, (float)p[c]);
    out[ch] = m;
}

// copy channels into f16 buffer; TS = source type (float or _Float16)
template <typename TS>
__global__ __launch_bounds__(256) void k_copy_chans(_Float16* __restrict__ dst, int ldd,
                                                    const TS* __restrict__ src, int lds_,
                                                    int C, int cols)
{
    int i = blockIdx.x * blockDim.x + threadIdx.x;
    if (i >= C * cols) return;
    int ch = i / cols, c = i % cols;
    dst[(size_t)ch * ldd + c] = (_Float16)(float)src[(size_t)ch * lds_ + c];
}

__global__ __launch_bounds__(256) void k_bcast_chans(_Float16* __restrict__ dst, int cols,
                                                     const float* __restrict__ vec, int C)
{
    int i = blockIdx.x * blockDim.x + threadIdx.x;
    if (i >= C * cols) return;
    int ch = i / cols, c = i % cols;
    dst[(size_t)ch * cols + c] = (_Float16)vec[ch];
}

__global__ __launch_bounds__(32) void k_copy_vec(float* __restrict__ dst,
                                                 const float* __restrict__ src, int n)
{
    int i = threadIdx.x;
    if (i < n) dst[i] = src[i];
}

// 3-NN interpolation; f16 feats, f32 math, f16 out. Strict < preserves top_k's
// lowest-index tie-break.
__global__ __launch_bounds__(256) void k_interp(const float* __restrict__ dxyz, int ldd, int Nd,
                                                const float* __restrict__ sxyz, int lds_, int Ns,
                                                const _Float16* __restrict__ feats, int ldf,
                                                int C, _Float16* __restrict__ out, int ldo)
{
    int i = blockIdx.x * blockDim.x + threadIdx.x;
    if (i >= Nd) return;
    float px = dxyz[i], py = dxyz[ldd + i], pz = dxyz[2 * ldd + i];
    float d0 = 3.4e38f, d1 = 3.4e38f, d2v = 3.4e38f;
    int i0 = 0, i1 = 0, i2 = 0;
    for (int s = 0; s < Ns; ++s) {
        float dx = sxyz[s] - px, dy = sxyz[lds_ + s] - py, dz = sxyz[2 * lds_ + s] - pz;
        float d = dx * dx + dy * dy + dz * dz;
        if (d < d0)       { d2v = d1; i2 = i1; d1 = d0; i1 = i0; d0 = d; i0 = s; }
        else if (d < d1)  { d2v = d1; i2 = i1; d1 = d;  i1 = s; }
        else if (d < d2v) { d2v = d;  i2 = s; }
    }
    float w0 = 1.0f / (fmaxf(d0, 0.0f) + 1e-8f);
    float w1 = 1.0f / (fmaxf(d1, 0.0f) + 1e-8f);
    float w2 = 1.0f / (fmaxf(d2v, 0.0f) + 1e-8f);
    float ws = w0 + w1 + w2;
    w0 /= ws; w1 /= ws; w2 /= ws;
    for (int ch = 0; ch < C; ++ch) {
        const _Float16* f = feats + (size_t)ch * ldf;
        out[(size_t)ch * ldo + i] =
            (_Float16)(w0 * (float)f[i0] + w1 * (float)f[i1] + w2 * (float)f[i2]);
    }
}

// ---------------------------------------------------------------------------
// Layer table: wi = d_in index of W (pytree order, sorted keys:
// cls,fp0,fp1,fp2,sa0,sa1,sa2; per layer W,b,beta,g).
struct LayerDef { int wi, co, cin, relu; };
static const LayerDef LD[18] = {
    {2, 128, 128, 1},   // 0  cls0
    {6, 2, 128, 0},     // 1  cls1 (plain, f32 out)
    {8, 256, 1296, 1},  // 2  fp0_0
    {12, 256, 256, 1},  // 3  fp0_1
    {16, 256, 384, 1},  // 4  fp1_0
    {20, 128, 256, 1},  // 5  fp1_1
    {24, 128, 134, 1},  // 6  fp2_0
    {28, 128, 128, 1},  // 7  fp2_1
    {32, 128, 128, 1},  // 8  fp2_2
    {36, 64, 6, 1},     // 9  sa0_0
    {40, 64, 64, 1},    // 10 sa0_1
    {44, 128, 64, 1},   // 11 sa0_2
    {48, 128, 131, 1},  // 12 sa1_0
    {52, 128, 128, 1},  // 13 sa1_1
    {56, 256, 128, 1},  // 14 sa1_2
    {60, 256, 259, 1},  // 15 sa2_0
    {64, 512, 256, 1},  // 16 sa2_1
    {68, 1024, 512, 1}, // 17 sa2_2
};

extern "C" void kernel_launch(void* const* d_in, const int* in_sizes, int n_in,
                              void* d_out, int out_size, void* d_ws, size_t ws_size,
                              hipStream_t stream)
{
    (void)in_sizes; (void)n_in; (void)out_size; (void)ws_size;
    const float* F  = (const float*)d_in[0];   // (16, 6, 8192)
    const float* OH = (const float*)d_in[1];   // (16, 16)
    auto P = [&](int i) { return (const float*)d_in[i]; };

    // padded parameter sizes
    int coP[18], cinP[18];
    size_t whOff[18], pvOff[18], whTot = 0, pvTot = 0;
    for (int i = 0; i < 18; ++i) {
        coP[i]  = (LD[i].co + 63) & ~63;
        cinP[i] = (LD[i].cin + 31) & ~31;
        pvOff[i] = pvTot; pvTot += (size_t)3 * coP[i];
        whOff[i] = whTot; whTot += (size_t)coP[i] * cinP[i];
    }

    // 16B-aligned bump allocator over d_ws
    char* wsb = (char*)d_ws;
    size_t off = 0;
    auto A16 = [&](size_t bytes) -> void* {
        off = (off + 15) & ~(size_t)15;
        void* p = wsb + off;
        off += bytes;
        return p;
    };
    _Float16* bufA = (_Float16*)A16((size_t)2097152 * 2);  // >= 160x8192, 64x32768, 1312x128
    _Float16* bufB = (_Float16*)A16((size_t)4194304 * 2);  // >= 128x32768
    _Float16* f1   = (_Float16*)A16((size_t)128 * 512 * 2);
    _Float16* f2   = (_Float16*)A16((size_t)256 * 128 * 2);
    _Float16* fp0o = (_Float16*)A16((size_t)256 * 128 * 2);
    _Float16* fp1o = (_Float16*)A16((size_t)128 * 512 * 2);
    float* gcat = (float*)A16(2048 * 4);          // 1024 global feat + 16 one-hot
    float* c1   = (float*)A16(3 * 512 * 4);
    float* c2   = (float*)A16(3 * 128 * 4);
    int* idx1  = (int*)A16(512 * 4);
    int* idx2  = (int*)A16(128 * 4);
    int* nidx1 = (int*)A16((size_t)512 * 64 * 4);
    int* nidx2 = (int*)A16((size_t)128 * 64 * 4);
    float* pvBase = (float*)A16(pvTot * 4);
    _Float16* whBase = (_Float16*)A16(whTot * 2);

    // one-time per-launch parameter prep
    for (int i = 0; i < 18; ++i) {
        int ne = coP[i] * cinP[i];
        k_cvt_w<<<(ne + 255) / 256, 256, 0, stream>>>(P(LD[i].wi), LD[i].co, LD[i].cin,
                                                      whBase + whOff[i], coP[i], cinP[i]);
        const float* g  = LD[i].relu ? P(LD[i].wi + 3) : nullptr;
        const float* be = LD[i].relu ? P(LD[i].wi + 2) : nullptr;
        k_pad_params<<<(coP[i] + 255) / 256, 256, 0, stream>>>(P(LD[i].wi + 1), g, be,
                                                               LD[i].co, coP[i],
                                                               pvBase + pvOff[i]);
    }

    auto G = [&](int li, const _Float16* X, _Float16* Y, int cols) {
        int tm = coP[li] / GEMM_BM, tn = cols / GEMM_BN;
        k_gemm_wmma<_Float16><<<tm * tn, 256, 0, stream>>>(
            whBase + whOff[li], pvBase + pvOff[li], X, Y, LD[li].co, coP[li], cinP[li],
            cols, LD[li].relu);
    };
    auto Z = [&](_Float16* p, int n) {  // zero-fill activation pad rows
        k_zeroh<<<(n + 255) / 256, 256, 0, stream>>>(p, n);
    };

    const int N = 8192;
    for (int b = 0; b < 16; ++b) {
        const float* Fb = F + (size_t)b * 6 * N;     // coords ch0..2, feats ch3..5
        // ---- SA0: m=512, r=0.2, K=64, MLP 6->64->64->128 ----
        k_fps<<<1, 256, 0, stream>>>(Fb, N, N, 512, idx1);
        k_gather_centers<<<6, 256, 0, stream>>>(Fb, N, idx1, 512, c1);
        k_ball<<<2, 256, 0, stream>>>(c1, 512, Fb, N, N, 0.04f, 64, nidx1);
        {
            int cols = 512 * 64;
            dim3 gr(cols / 256, 6);
            k_gather_sa<float><<<gr, 256, 0, stream>>>(bufA, cols, 64, nidx1, Fb, N, c1,
                                                       512, Fb + 3 * (size_t)N, N);
            Z(bufA + (size_t)6 * cols, (cinP[9] - 6) * cols);
            G(9, bufA, bufB, cols);
            G(10, bufB, bufA, cols);
            G(11, bufA, bufB, cols);
            k_maxpool_k<<<(128 * 512 + 255) / 256, 256, 0, stream>>>(bufB, 512, 64, 128, f1);
        }
        // ---- SA1: m=128, r=0.4, K=64, MLP 131->128->128->256 ----
        k_fps<<<1, 256, 0, stream>>>(c1, 512, 512, 128, idx2);
        k_gather_centers<<<2, 256, 0, stream>>>(c1, 512, idx2, 128, c2);
        k_ball<<<1, 256, 0, stream>>>(c2, 128, c1, 512, 512, 0.16f, 64, nidx2);
        {
            int cols = 128 * 64;
            dim3 gr(cols / 256, 131);
            k_gather_sa<_Float16><<<gr, 256, 0, stream>>>(bufA, cols, 64, nidx2, c1, 512,
                                                          c2, 128, f1, 512);
            Z(bufA + (size_t)131 * cols, (cinP[12] - 131) * cols);
            G(12, bufA, bufB, cols);
            G(13, bufB, bufA, cols);
            G(14, bufA, bufB, cols);
            k_maxpool_k<<<(256 * 128 + 255) / 256, 256, 0, stream>>>(bufB, 128, 64, 256, f2);
        }
        // ---- SA2 (global): MLP 259->256->512->1024, max over 128 cols ----
        {
            int cols = 128;
            k_copy_chans<_Float16><<<(256 * cols + 255) / 256, 256, 0, stream>>>(
                bufA, cols, f2, 128, 256, cols);
            k_copy_chans<float><<<2, 256, 0, stream>>>(bufA + (size_t)256 * cols, cols, c2,
                                                       128, 3, cols);
            Z(bufA + (size_t)259 * cols, (cinP[15] - 259) * cols);
            G(15, bufA, bufB, cols);
            G(16, bufB, bufA, cols);
            G(17, bufA, bufB, cols);
            k_maxcols<<<4, 256, 0, stream>>>(bufB, 1024, cols, gcat);
            k_copy_vec<<<1, 32, 0, stream>>>(gcat + 1024, OH + b * 16, 16);
        }
        // ---- FP0: broadcast (sparse = 1 pt) + skip f2; MLP 1296->256->256 ----
        {
            int cols = 128;
            k_bcast_chans<<<(1040 * cols + 255) / 256, 256, 0, stream>>>(bufA, cols, gcat,
                                                                         1040);
            k_copy_chans<_Float16><<<(256 * cols + 255) / 256, 256, 0, stream>>>(
                bufA + (size_t)1040 * cols, cols, f2, 128, 256, cols);
            Z(bufA + (size_t)1296 * cols, (cinP[2] - 1296) * cols);
            G(2, bufA, bufB, cols);
            G(3, bufB, fp0o, cols);
        }
        // ---- FP1: interp c2->c1 + skip f1; MLP 384->256->128 ----
        {
            int cols = 512;
            k_interp<<<2, 256, 0, stream>>>(c1, 512, 512, c2, 128, 128, fp0o, 128, 256,
                                            bufA, cols);
            k_copy_chans<_Float16><<<(128 * cols + 255) / 256, 256, 0, stream>>>(
                bufA + (size_t)256 * cols, cols, f1, 512, 128, cols);
            G(4, bufA, bufB, cols);
            G(5, bufB, fp1o, cols);
        }
        // ---- FP2 + CLS: interp c1->c0 + skip raw features; 134->128->128->128,
        //      cls 128->128 (relu) -> 2 (plain, f32) straight into d_out ----
        {
            int cols = N;
            k_interp<<<N / 256, 256, 0, stream>>>(Fb, N, N, c1, 512, 512, fp1o, 512, 128,
                                                  bufA, cols);
            k_copy_chans<float><<<(6 * cols + 255) / 256, 256, 0, stream>>>(
                bufA + (size_t)128 * cols, cols, Fb, N, 6, cols);
            Z(bufA + (size_t)134 * cols, (cinP[6] - 134) * cols);
            G(6, bufA, bufB, cols);
            G(7, bufB, bufA, cols);
            G(8, bufA, bufB, cols);
            G(0, bufB, bufA, cols);
            float* outb = (float*)d_out + (size_t)b * 2 * N;
            k_gemm_wmma<float><<<(coP[1] / GEMM_BM) * (cols / GEMM_BN), 256, 0, stream>>>(
                whBase + whOff[1], pvBase + pvOff[1], bufA, outb, LD[1].co, coP[1],
                cinP[1], cols, LD[1].relu);
        }
    }
}
